// EGNNLayer_11716670783711
// MI455X (gfx1250) — matrix operand order repaired and checked by reading, not compiled, not run on hardware
//
#include <hip/hip_runtime.h>
#include <cstddef>

#define NN 50000
#define EE 800000
#define HH 128

typedef unsigned int u32;
typedef unsigned short u16;
typedef __attribute__((ext_vector_type(4)))  u32    u32x4;
typedef __attribute__((ext_vector_type(8)))  float  v8f;
typedef __attribute__((ext_vector_type(16))) __bf16 v16bf;

typedef union { u32x4 q[2]; v16bf v; } V16;

static __device__ __forceinline__ u16 f2bf(float f){
  u32 u = __builtin_bit_cast(u32, f);
  u32 r = u + 0x7FFFu + ((u >> 16) & 1u);
  return (u16)(r >> 16);
}
static __device__ __forceinline__ float silu(float v){
  return v / (1.0f + __expf(-v));
}
static __device__ __forceinline__ v8f vzero(){
  v8f z;
#pragma unroll
  for (int i = 0; i < 8; ++i) z[i] = 0.f;
  return z;
}
// A fragment: 16x32 bf16. lane<16: row=lane, K slots {0..7,16..23}; lane>=16: row=lane-16, K {8..15,24..31}.
static __device__ __forceinline__ v16bf ldA(const u16* p){
  V16 t; t.q[0] = *(const u32x4*)p; t.q[1] = *(const u32x4*)(p + 16); return t.v;
}
// B fragment buffer: [kt][nt][lane][16 bf16] contiguous per lane.
static __device__ __forceinline__ v16bf ldB(const u16* base, int kt, int nt, int lane){
  const u32x4* q = (const u32x4*)(base + ((size_t)(((kt << 3) + nt) << 5) + lane) * 16);
  V16 t; t.q[0] = q[0]; t.q[1] = q[1]; return t.v;
}

__global__ void zero_f32(float* __restrict__ p, long long n){
  long long i = (long long)blockIdx.x * blockDim.x + threadIdx.x;
  long long st = (long long)gridDim.x * blockDim.x;
  for (; i < n; i += st) p[i] = 0.f;
}

__global__ void h_to_bf16(const float* __restrict__ h, u16* __restrict__ hb, long long n){
  long long i = (long long)blockIdx.x * blockDim.x + threadIdx.x;
  long long st = (long long)gridDim.x * blockDim.x;
  for (; i < n; i += st) hb[i] = f2bf(h[i]);
}

// Pack row-major W[K][128] into B-fragment order (bf16), zero-padding K >= Kact.
__global__ void pack_bfrag(const float* __restrict__ W, u16* __restrict__ out, int Kact, int KT){
  int idx = blockIdx.x * blockDim.x + threadIdx.x;
  int total = KT * 8 * 32 * 16;
  if (idx >= total) return;
  int slot = idx & 15;
  int ln   = (idx >> 4) & 31;
  int nt   = (idx >> 9) & 7;
  int kt   = idx >> 12;
  int k = kt * 32 + ((slot < 8) ? ((ln >> 4) * 8 + slot) : (16 + (ln >> 4) * 8 + (slot - 8)));
  int n = nt * 16 + (ln & 15);
  float v = (k < Kact) ? W[(size_t)k * 128 + n] : 0.f;
  out[idx] = f2bf(v);
}

__global__ __launch_bounds__(128) void egnn_edge(
    const int* __restrict__ ei, const float* __restrict__ x,
    const float* __restrict__ eattr, const u16* __restrict__ hb,
    const u16* __restrict__ we1f, const u16* __restrict__ we2f,
    const u16* __restrict__ wc1f,
    const float* __restrict__ we_b1, const float* __restrict__ we_b2,
    const float* __restrict__ wc_b1, const float* __restrict__ wc_w2,
    float* __restrict__ m_i, float* __restrict__ x_upd, float* __restrict__ dg)
{
  __shared__ alignas(16) u16 sA[4][16][296];   // 16x288 edge-input tile (bf16), padded stride
  __shared__ alignas(16) u16 sM[4][16][136];   // 16x128 activation tile (bf16)
  __shared__ float sCD[4][16][4];
  __shared__ int   sRC[4][2][16];
  __shared__ float sCW[4][16];

  const int w    = threadIdx.x >> 5;
  const int lane = threadIdx.x & 31;
  const int tile = blockIdx.x * 4 + w;
  if (tile * 16 >= EE) return;
  const int colr = lane & 15;
  const int half = lane >> 4;

  if (lane < 16){
    int e = tile * 16 + lane;
    int r = ei[e], c = ei[EE + e];
    sRC[w][0][lane] = r; sRC[w][1][lane] = c;
    float dx = x[3*r+0] - x[3*c+0];
    float dy = x[3*r+1] - x[3*c+1];
    float dz = x[3*r+2] - x[3*c+2];
    float d2 = dx*dx + dy*dy + dz*dz;
    sCD[w][lane][0] = dx; sCD[w][lane][1] = dy; sCD[w][lane][2] = dz; sCD[w][lane][3] = d2;
    sA[w][lane][256] = f2bf(d2);
#pragma unroll
    for (int j = 0; j < 16; ++j) sA[w][lane][257+j] = f2bf(eattr[(size_t)e*16 + j]);
#pragma unroll
    for (int j = 273; j < 288; ++j) sA[w][lane][j] = 0;
  }
  __builtin_amdgcn_wave_barrier();

  { // gather h[row], h[col] (bf16) -> cols [0,128) and [128,256); 2 lanes per edge
    int m2 = lane >> 1, p = lane & 1;
    int er = sRC[w][0][m2], ec = sRC[w][1][m2];
    const u32x4* hr = (const u32x4*)(hb + (size_t)er*HH + p*64);
    const u32x4* hc = (const u32x4*)(hb + (size_t)ec*HH + p*64);
    u32x4* d0 = (u32x4*)&sA[w][m2][p*64];
    u32x4* d1 = (u32x4*)&sA[w][m2][128 + p*64];
#pragma unroll
    for (int i = 0; i < 8; ++i){ d0[i] = hr[i]; d1[i] = hc[i]; }
  }
  __builtin_amdgcn_wave_barrier();

  v8f acc[8];
  // ---- edge MLP layer 1: [16x288] @ [288x128] ----
#pragma unroll
  for (int nt = 0; nt < 8; ++nt) acc[nt] = vzero();
#pragma unroll
  for (int kt = 0; kt < 9; ++kt){
    v16bf a = ldA(&sA[w][colr][kt*32 + half*8]);
#pragma unroll
    for (int nt = 0; nt < 8; ++nt){
      v16bf b = ldB(we1f, kt, nt, lane);
      acc[nt] = __builtin_amdgcn_wmma_f32_16x16x32_bf16(false, a, false, b, (short)0, acc[nt], false, false);
    }
  }
#pragma unroll
  for (int nt = 0; nt < 8; ++nt){
    float b1 = we_b1[nt*16 + colr];
#pragma unroll
    for (int r = 0; r < 8; ++r)
      sM[w][r + 8*half][nt*16 + colr] = f2bf(silu(acc[nt][r] + b1));
  }
  __builtin_amdgcn_wave_barrier();

  // ---- edge MLP layer 2: [16x128] @ [128x128] ----
#pragma unroll
  for (int nt = 0; nt < 8; ++nt) acc[nt] = vzero();
#pragma unroll
  for (int kt = 0; kt < 4; ++kt){
    v16bf a = ldA(&sM[w][colr][kt*32 + half*8]);
#pragma unroll
    for (int nt = 0; nt < 8; ++nt){
      v16bf b = ldB(we2f, kt, nt, lane);
      acc[nt] = __builtin_amdgcn_wmma_f32_16x16x32_bf16(false, a, false, b, (short)0, acc[nt], false, false);
    }
  }
  int rows[8];
#pragma unroll
  for (int r = 0; r < 8; ++r) rows[r] = sRC[w][0][r + 8*half];
#pragma unroll
  for (int nt = 0; nt < 8; ++nt){
    float b2 = we_b2[nt*16 + colr];
#pragma unroll
    for (int r = 0; r < 8; ++r){
      float v = silu(acc[nt][r] + b2);                       // m_ij element (M=r+8*half, N=nt*16+colr)
      sM[w][r + 8*half][nt*16 + colr] = f2bf(v);
      atomicAdd(&m_i[(size_t)rows[r]*HH + nt*16 + colr], v); // segment-sum scatter (L2-resident)
    }
  }
  __builtin_amdgcn_wave_barrier();

  // ---- coord MLP layer 1: silu(m_ij @ wc_w1 + b) ----
#pragma unroll
  for (int nt = 0; nt < 8; ++nt) acc[nt] = vzero();
#pragma unroll
  for (int kt = 0; kt < 4; ++kt){
    v16bf a = ldA(&sM[w][colr][kt*32 + half*8]);
#pragma unroll
    for (int nt = 0; nt < 8; ++nt){
      v16bf b = ldB(wc1f, kt, nt, lane);
      acc[nt] = __builtin_amdgcn_wmma_f32_16x16x32_bf16(false, a, false, b, (short)0, acc[nt], false, false);
    }
  }
  // dot with wc_w2[128] per edge, then tanh
  float part[8];
#pragma unroll
  for (int r = 0; r < 8; ++r) part[r] = 0.f;
#pragma unroll
  for (int nt = 0; nt < 8; ++nt){
    float bc = wc_b1[nt*16 + colr];
    float w2 = wc_w2[nt*16 + colr];
#pragma unroll
    for (int r = 0; r < 8; ++r) part[r] += silu(acc[nt][r] + bc) * w2;
  }
#pragma unroll
  for (int r = 0; r < 8; ++r){
    float s = part[r];
    s += __shfl_xor(s, 1, 32);
    s += __shfl_xor(s, 2, 32);
    s += __shfl_xor(s, 4, 32);
    s += __shfl_xor(s, 8, 32);
    if (colr == 0) sCW[w][r + 8*half] = tanhf(s);
  }
  __builtin_amdgcn_wave_barrier();

  if (lane < 16){
    int er = sRC[w][0][lane];
    float d2 = sCD[w][lane][3];
    float sc = sCW[w][lane] / sqrtf(d2 + 1e-8f);
    atomicAdd(&x_upd[(size_t)er*3 + 0], sCD[w][lane][0] * sc);
    atomicAdd(&x_upd[(size_t)er*3 + 1], sCD[w][lane][1] * sc);
    atomicAdd(&x_upd[(size_t)er*3 + 2], sCD[w][lane][2] * sc);
    atomicAdd(&dg[er], 1.0f);
  }
}

__global__ __launch_bounds__(128) void egnn_node(
    const u16* __restrict__ hb, const float* __restrict__ m_i,
    const float* __restrict__ h,
    const u16* __restrict__ wn1f, const u16* __restrict__ wn2f,
    const float* __restrict__ wn_b1, const float* __restrict__ wn_b2,
    const float* __restrict__ ln_g, const float* __restrict__ ln_b,
    float* __restrict__ hout)
{
  __shared__ alignas(16) u16 sA[4][16][264];   // 16x256 [h | m_i] bf16
  __shared__ alignas(16) u16 sM[4][16][136];
  const int w = threadIdx.x >> 5, lane = threadIdx.x & 31;
  const int tile = blockIdx.x * 4 + w;
  if (tile * 16 >= NN) return;
  const int colr = lane & 15, half = lane >> 4;
  const int n0 = tile * 16;

  {
    int m2 = lane >> 1, p = lane & 1;
    int nd = n0 + m2;
    const u32x4* hr = (const u32x4*)(hb + (size_t)nd*HH + p*64);
    u32x4* d0 = (u32x4*)&sA[w][m2][p*64];
#pragma unroll
    for (int i = 0; i < 8; ++i) d0[i] = hr[i];
    const float* mi = m_i + (size_t)nd*HH + p*64;
#pragma unroll
    for (int j = 0; j < 64; ++j) sA[w][m2][128 + p*64 + j] = f2bf(mi[j]);
  }
  __builtin_amdgcn_wave_barrier();

  v8f acc[8];
#pragma unroll
  for (int nt = 0; nt < 8; ++nt) acc[nt] = vzero();
#pragma unroll
  for (int kt = 0; kt < 8; ++kt){
    v16bf a = ldA(&sA[w][colr][kt*32 + half*8]);
#pragma unroll
    for (int nt = 0; nt < 8; ++nt){
      v16bf b = ldB(wn1f, kt, nt, lane);
      acc[nt] = __builtin_amdgcn_wmma_f32_16x16x32_bf16(false, a, false, b, (short)0, acc[nt], false, false);
    }
  }
#pragma unroll
  for (int nt = 0; nt < 8; ++nt){
    float b1 = wn_b1[nt*16 + colr];
#pragma unroll
    for (int r = 0; r < 8; ++r)
      sM[w][r + 8*half][nt*16 + colr] = f2bf(silu(acc[nt][r] + b1));
  }
  __builtin_amdgcn_wave_barrier();
#pragma unroll
  for (int nt = 0; nt < 8; ++nt) acc[nt] = vzero();
#pragma unroll
  for (int kt = 0; kt < 4; ++kt){
    v16bf a = ldA(&sM[w][colr][kt*32 + half*8]);
#pragma unroll
    for (int nt = 0; nt < 8; ++nt){
      v16bf b = ldB(wn2f, kt, nt, lane);
      acc[nt] = __builtin_amdgcn_wmma_f32_16x16x32_bf16(false, a, false, b, (short)0, acc[nt], false, false);
    }
  }
  // residual + layernorm (f32)
  float rv[8][8];
#pragma unroll
  for (int nt = 0; nt < 8; ++nt){
    float b2 = wn_b2[nt*16 + colr];
#pragma unroll
    for (int r = 0; r < 8; ++r){
      int M = r + 8*half;
      float hv = h[(size_t)(n0 + M)*HH + nt*16 + colr];
      rv[nt][r] = hv + acc[nt][r] + b2;
    }
  }
  float s1[8], s2[8];
#pragma unroll
  for (int r = 0; r < 8; ++r){ s1[r] = 0.f; s2[r] = 0.f; }
#pragma unroll
  for (int nt = 0; nt < 8; ++nt)
#pragma unroll
    for (int r = 0; r < 8; ++r){ s1[r] += rv[nt][r]; s2[r] += rv[nt][r]*rv[nt][r]; }
#pragma unroll
  for (int r = 0; r < 8; ++r){
    float a = s1[r], b = s2[r];
    a += __shfl_xor(a, 1, 32); b += __shfl_xor(b, 1, 32);
    a += __shfl_xor(a, 2, 32); b += __shfl_xor(b, 2, 32);
    a += __shfl_xor(a, 4, 32); b += __shfl_xor(b, 4, 32);
    a += __shfl_xor(a, 8, 32); b += __shfl_xor(b, 8, 32);
    s1[r] = a * (1.f/HH);
    s2[r] = b * (1.f/HH) - s1[r]*s1[r];
  }
#pragma unroll
  for (int nt = 0; nt < 8; ++nt){
    float g = ln_g[nt*16 + colr], bb = ln_b[nt*16 + colr];
#pragma unroll
    for (int r = 0; r < 8; ++r){
      int M = r + 8*half;
      float inv = rsqrtf(s2[r] + 1e-5f);
      hout[(size_t)(n0 + M)*HH + nt*16 + colr] = (rv[nt][r] - s1[r]) * inv * g + bb;
    }
  }
}

__global__ void x_out_kernel(const float* __restrict__ x, const float* __restrict__ xu,
                             const float* __restrict__ dg, float* __restrict__ out, int n3){
  int i = blockIdx.x * blockDim.x + threadIdx.x;
  if (i >= n3) return;
  int nnode = i / 3;
  float d = dg[nnode];
  d = d < 1.f ? 1.f : d;
  out[i] = x[i] + xu[i] / d;
}

extern "C" void kernel_launch(void* const* d_in, const int* in_sizes, int n_in,
                              void* d_out, int out_size, void* d_ws, size_t ws_size,
                              hipStream_t stream) {
  (void)in_sizes; (void)n_in; (void)out_size; (void)ws_size;
  const float* h     = (const float*)d_in[0];
  const float* x     = (const float*)d_in[1];
  const float* eattr = (const float*)d_in[2];
  const float* we_w1 = (const float*)d_in[3];
  const float* we_b1 = (const float*)d_in[4];
  const float* we_w2 = (const float*)d_in[5];
  const float* we_b2 = (const float*)d_in[6];
  const float* wc_w1 = (const float*)d_in[7];
  const float* wc_b1 = (const float*)d_in[8];
  const float* wc_w2 = (const float*)d_in[9];
  const float* wn_w1 = (const float*)d_in[10];
  const float* wn_b1 = (const float*)d_in[11];
  const float* wn_w2 = (const float*)d_in[12];
  const float* wn_b2 = (const float*)d_in[13];
  const float* ln_g  = (const float*)d_in[14];
  const float* ln_b  = (const float*)d_in[15];
  const int*   ei    = (const int*)d_in[16];
  float* out = (float*)d_out;

  // workspace layout
  float* m_i   = (float*)d_ws;                 // N*H f32
  float* x_upd = m_i + (size_t)NN*HH;          // N*3 f32
  float* dg    = x_upd + (size_t)NN*3;         // N f32
  u16*  hb     = (u16*)(dg + NN);              // N*H bf16
  u16*  we1f   = hb + (size_t)NN*HH;           // 9 ktiles
  u16*  we2f   = we1f + 9*8*512;
  u16*  wc1f   = we2f + 4*8*512;
  u16*  wn1f   = wc1f + 4*8*512;
  u16*  wn2f   = wn1f + 8*8*512;

  zero_f32<<<1024, 256, 0, stream>>>(m_i, (long long)NN*(HH+4));
  h_to_bf16<<<1024, 256, 0, stream>>>(h, hb, (long long)NN*HH);
  pack_bfrag<<<(9*4096 + 255)/256, 256, 0, stream>>>(we_w1, we1f, 273, 9);
  pack_bfrag<<<(4*4096 + 255)/256, 256, 0, stream>>>(we_w2, we2f, 128, 4);
  pack_bfrag<<<(4*4096 + 255)/256, 256, 0, stream>>>(wc_w1, wc1f, 128, 4);
  pack_bfrag<<<(8*4096 + 255)/256, 256, 0, stream>>>(wn_w1, wn1f, 256, 8);
  pack_bfrag<<<(4*4096 + 255)/256, 256, 0, stream>>>(wn_w2, wn2f, 128, 4);

  egnn_edge<<<EE/64, 128, 0, stream>>>(ei, x, eattr, hb, we1f, we2f, wc1f,
                                       we_b1, we_b2, wc_b1, wc_w2,
                                       m_i, x_upd, dg);
  egnn_node<<<(NN/16 + 3)/4, 128, 0, stream>>>(hb, m_i, h, wn1f, wn2f,
                                               wn_b1, wn_b2, ln_g, ln_b, out);
  x_out_kernel<<<(NN*3 + 255)/256, 256, 0, stream>>>(x, x_upd, dg, out + (size_t)NN*HH, NN*3);
}
